// spiking_coESN_rescaled_17600775979595
// MI455X (gfx1250) — compile-verified, matrix operand order checked by reading, not run on hardware
//
#include <hip/hip_runtime.h>
#include <hip/hip_bf16.h>

typedef __attribute__((ext_vector_type(16))) _Float16 v16h;
typedef __attribute__((ext_vector_type(8)))  _Float16 v8h;
typedef __attribute__((ext_vector_type(8)))  float    v8f;

union Frag16 { v16h v; v8h h[2]; };

#define N_HID   1024
#define LSEQ    512
#define BATCH   128
#define ROWH    1032          // padded LDS row stride in halfs (bank-conflict pad)
#define THREADS 512
#define TILES   4             // 16-col WMMA tiles per wave (16 waves * 64 cols = 1024)

#define DTC     0.05f
#define KMEM    0.9975f                 // 1 - DT/LIF_TAU_M
#define RDECAY  0.81873075307798182f    // exp(-DT/TAU_REF)

#define SMEM_BYTES (16*ROWH*2*2 + 16*LSEQ*4 + 32)

// ---- kernel 0: f32->f16 transpose of both weight matrices + zero accumulators ----
__global__ void coesn_convert_kernel(const float* __restrict__ h2h,
                                     const float* __restrict__ lif2hrf,
                                     _Float16* __restrict__ bh,
                                     _Float16* __restrict__ bl,
                                     float* __restrict__ acc) {
  size_t i = (size_t)blockIdx.x * blockDim.x + threadIdx.x;
  if (i < 8) acc[i] = 0.0f;
  if (i < (size_t)N_HID * N_HID) {
    size_t k = i >> 10, n = i & (N_HID - 1);
    bh[n * (size_t)N_HID + k] = (_Float16)h2h[i];          // column-major for B frags
  } else if (i < 2ull * N_HID * N_HID) {
    size_t j = i - (size_t)N_HID * N_HID;
    size_t k = j >> 10, n = j & (N_HID - 1);
    bl[n * (size_t)N_HID + k] = (_Float16)lif2hrf[j];
  }
}

// ---- kernel 1: persistent per-batch-tile scan, state lives in WMMA C-layout VGPRs ----
__global__ __launch_bounds__(THREADS, 1)
void coesn_scan_kernel(const float* __restrict__ x,
                       const float* __restrict__ x2h,
                       const float* __restrict__ bias,
                       const float* __restrict__ gamma,
                       const float* __restrict__ epsilon,
                       const float* __restrict__ spike_gain,
                       const _Float16* __restrict__ Bh,
                       const _Float16* __restrict__ Bl,
                       float* __restrict__ out_hy,
                       float* __restrict__ acc) {
  extern __shared__ char smem[];
  _Float16* As = (_Float16*)smem;               // 16 x ROWH  : hrf spikes (A of GEMM1)
  _Float16* Al = As + 16 * ROWH;                // 16 x ROWH  : lif spikes (A of GEMM2)
  float*    xs = (float*)(Al + 16 * ROWH);      // 16 x LSEQ  : input slice
  float*    red = xs + 16 * LSEQ;               // 8 floats   : stat reduction

  const int tid   = threadIdx.x;
  const int lane  = tid & 31;
  const int wv    = tid >> 5;
  const int nlo   = lane & 15;
  const int hsel  = lane >> 4;       // half-wave selector (WMMA K/M split)
  const int rbase = hsel * 8;        // C/D layout: lanes 16-31 hold M=8..15
  const int bb    = blockIdx.x * 16; // batch tile base row

  for (int i = tid; i < 16 * ROWH; i += THREADS) As[i] = (_Float16)0.0f; // s0 = 0
  for (int i = tid; i < 16 * LSEQ; i += THREADS) {
    int m = i >> 9, t = i & (LSEQ - 1);
    xs[i] = x[(size_t)(bb + m) * LSEQ + t];     // x is (B, L, 1)
  }
  if (tid < 8) red[tid] = 0.0f;

  int   col[TILES];
  float biasC[TILES], gamC[TILES], epsC[TILES], x2hC[TILES];
#pragma unroll
  for (int tI = 0; tI < TILES; ++tI) {
    col[tI]   = wv * (TILES * 16) + tI * 16 + nlo;
    biasC[tI] = bias[col[tI]];
    gamC[tI]  = gamma[col[tI]];
    epsC[tI]  = epsilon[col[tI]];
    x2hC[tI]  = x2h[col[tI]];
  }
  const float g = spike_gain[0];

  float hy[TILES][8], hz[TILES][8], rf[TILES][8], lv[TILES][8];
#pragma unroll
  for (int tI = 0; tI < TILES; ++tI)
#pragma unroll
    for (int r = 0; r < 8; ++r) { hy[tI][r] = 0.f; hz[tI][r] = 0.f; rf[tI][r] = 0.f; lv[tI][r] = 0.f; }

  float a_thrf = 0.f, a_tlif = 0.f, a_vs = 0.f, a_vsq = 0.f, a_ls = 0.f, a_lsq = 0.f;

  __syncthreads();

  const _Float16* arowS = As + nlo * ROWH;   // A frag: M = lane&15
  const _Float16* arowL = Al + nlo * ROWH;

  for (int t = 0; t < LSEQ; ++t) {
    float xv[8];
#pragma unroll
    for (int r = 0; r < 8; ++r) xv[r] = xs[(rbase + r) * LSEQ + t];

    // ---------- GEMM1: cur = s @ h2h + x*x2h + bias ----------
    v8f c[TILES];
#pragma unroll
    for (int tI = 0; tI < TILES; ++tI)
#pragma unroll
      for (int r = 0; r < 8; ++r) c[tI][r] = biasC[tI] + xv[r] * x2hC[tI];

    for (int kt = 0; kt < N_HID / 32; ++kt) {
      const int ka = kt * 32 + hsel * 8;       // 16-bit A layout K split per half-wave
      Frag16 a;
      a.h[0] = *(const v8h*)(arowS + ka);
      a.h[1] = *(const v8h*)(arowS + ka + 16);
      __builtin_prefetch((const void*)(Bh + (size_t)col[0] * N_HID + ka + 32), 0, 3);
#pragma unroll
      for (int tI = 0; tI < TILES; ++tI) {
        const _Float16* bp = Bh + (size_t)col[tI] * N_HID + ka;
        Frag16 b;
        b.h[0] = *(const v8h*)(bp);
        b.h[1] = *(const v8h*)(bp + 16);
        c[tI] = __builtin_amdgcn_wmma_f32_16x16x32_f16(false, a.v, false, b.v,
                                                       (short)0, c[tI], false, false);
      }
    }

    // ---------- LIF state update, emit lif spikes ----------
#pragma unroll
    for (int tI = 0; tI < TILES; ++tI)
#pragma unroll
      for (int r = 0; r < 8; ++r) {
        float v  = lv[tI][r] * KMEM + DTC * c[tI][r];
        float sp = (v > 1.0f) ? 1.0f : 0.0f;   // THETA_LIF = 1
        v -= sp;
        lv[tI][r] = v;
        Al[(rbase + r) * ROWH + col[tI]] = (_Float16)sp;
        a_tlif += sp; a_vs += v; a_vsq += v * v;
      }
    __syncthreads();

    // ---------- GEMM2: l2h = lif_s @ lif2hrf ----------
#pragma unroll
    for (int tI = 0; tI < TILES; ++tI)
#pragma unroll
      for (int r = 0; r < 8; ++r) c[tI][r] = 0.0f;

    for (int kt = 0; kt < N_HID / 32; ++kt) {
      const int ka = kt * 32 + hsel * 8;
      Frag16 a;
      a.h[0] = *(const v8h*)(arowL + ka);
      a.h[1] = *(const v8h*)(arowL + ka + 16);
      __builtin_prefetch((const void*)(Bl + (size_t)col[0] * N_HID + ka + 32), 0, 3);
#pragma unroll
      for (int tI = 0; tI < TILES; ++tI) {
        const _Float16* bp = Bl + (size_t)col[tI] * N_HID + ka;
        Frag16 b;
        b.h[0] = *(const v8h*)(bp);
        b.h[1] = *(const v8h*)(bp + 16);
        c[tI] = __builtin_amdgcn_wmma_f32_16x16x32_f16(false, a.v, false, b.v,
                                                       (short)0, c[tI], false, false);
      }
    }

    // ---------- HRF state update, emit hrf spikes ----------
#pragma unroll
    for (int tI = 0; tI < TILES; ++tI)
#pragma unroll
      for (int r = 0; r < 8; ++r) {
        float l2 = c[tI][r];
        a_ls += l2; a_lsq += l2 * l2;
        float z = hz[tI][r], y = hy[tI][r];
        z += DTC * (g * l2 - gamC[tI] * y - epsC[tI] * z);
        y += DTC * z;
        float sp = ((y - 1.0f - rf[tI][r]) > 0.0f) ? 1.0f : 0.0f; // THETA_RF = 1
        rf[tI][r] = rf[tI][r] * RDECAY + sp;
        hz[tI][r] = z; hy[tI][r] = y;
        a_thrf += sp;
        As[(rbase + r) * ROWH + col[tI]] = (_Float16)sp;
      }
    __syncthreads();
  }

  // ---------- write hy ----------
#pragma unroll
  for (int tI = 0; tI < TILES; ++tI)
#pragma unroll
    for (int r = 0; r < 8; ++r)
      out_hy[(size_t)(bb + rbase + r) * N_HID + col[tI]] = hy[tI][r];

  // ---------- scalar statistics reduction ----------
  atomicAdd(&red[0], a_thrf);
  atomicAdd(&red[1], a_tlif);
  atomicAdd(&red[2], a_vs);
  atomicAdd(&red[3], a_vsq);
  atomicAdd(&red[4], a_ls);
  atomicAdd(&red[5], a_lsq);
  __syncthreads();
  if (tid < 6) atomicAdd(&acc[tid], red[tid]);
}

// ---- kernel 2: finalize the 7 scalar outputs ----
__global__ void coesn_finalize_kernel(const float* __restrict__ acc,
                                      float* __restrict__ outs) {
  const float denom = (float)BATCH * (float)LSEQ * (float)N_HID;
  float r_hrf = acc[0] / denom;
  float r_lif = acc[1] / denom;
  float vm    = acc[2] / denom;
  float vstd  = sqrtf(fmaxf(acc[3] / denom - vm * vm, 0.0f));
  float lm    = acc[4] / denom;
  float lstd  = sqrtf(fmaxf(acc[5] / denom - lm * lm, 0.0f));
  outs[0] = r_hrf;  // r_total == r_hrf (count_lif_spikes=False)
  outs[1] = r_hrf;
  outs[2] = r_lif;
  outs[3] = vm;
  outs[4] = vstd;
  outs[5] = lm;
  outs[6] = lstd;
}

extern "C" void kernel_launch(void* const* d_in, const int* in_sizes, int n_in,
                              void* d_out, int out_size, void* d_ws, size_t ws_size,
                              hipStream_t stream) {
  const float* x        = (const float*)d_in[0];  // (128,512,1)
  const float* x2h      = (const float*)d_in[1];  // (1,1024)
  const float* h2h      = (const float*)d_in[2];  // (1024,1024)
  const float* bias     = (const float*)d_in[3];  // (1024,)
  const float* lif2hrf  = (const float*)d_in[4];  // (1024,1024)
  const float* gamma    = (const float*)d_in[5];  // (1024,)
  const float* epsilon  = (const float*)d_in[6];  // (1024,)
  const float* sgain    = (const float*)d_in[7];  // scalar

  float* out_hy   = (float*)d_out;                       // first 128*1024 floats
  float* out_scal = (float*)d_out + (size_t)BATCH * N_HID;

  _Float16* bh  = (_Float16*)d_ws;                       // 2 MB: h2h^T f16
  _Float16* bl  = bh + (size_t)N_HID * N_HID;            // 2 MB: lif2hrf^T f16
  float*    acc = (float*)(bl + (size_t)N_HID * N_HID);  // 8 f32 stat accumulators

  const int convTotal  = 2 * N_HID * N_HID;
  coesn_convert_kernel<<<convTotal / 256, 256, 0, stream>>>(h2h, lif2hrf, bh, bl, acc);

  coesn_scan_kernel<<<BATCH / 16, THREADS, SMEM_BYTES, stream>>>(
      x, x2h, bias, gamma, epsilon, sgain, bh, bl, out_hy, acc);

  coesn_finalize_kernel<<<1, 1, 0, stream>>>(acc, out_scal);
}
